// OpenMaxSeparation_16887811408235
// MI455X (gfx1250) — matrix operand order repaired and checked by reading, not compiled, not run on hardware
//
#include <hip/hip_runtime.h>
#include <math.h>

// ---------------------------------------------------------------------------
// Problem constants (fixed by the reference setup)
// ---------------------------------------------------------------------------
#define BDIM 16384   // batch
#define DDIM 2048    // feature dim (K)
#define CDIM 1000    // classes (N)

// ---------------------------------------------------------------------------
// CDNA5 WMMA vector types (wave32: 16x16x32 bf16 -> A/B = 16 bf16/lane,
// C/D = 8 f32/lane)
// ---------------------------------------------------------------------------
typedef __attribute__((ext_vector_type(16))) __bf16 v16bf;
typedef __attribute__((ext_vector_type(8)))  float  v8f;

union V16 {
  v16bf v;
  unsigned short s[16];
  uint4 q[2];
};

// float -> bf16 round-to-nearest-even via integer math (no __bf16 arithmetic)
__device__ __forceinline__ unsigned short f2bf_rne(float f) {
  unsigned int u = __float_as_uint(f);
  unsigned int r = u + 0x7FFFu + ((u >> 16) & 1u);
  return (unsigned short)(r >> 16);
}
__device__ __forceinline__ float bf2f(unsigned short h) {
  return __uint_as_float(((unsigned int)h) << 16);
}

// ---------------------------------------------------------------------------
// Pre-split centers into bf16 hi/lo planes (once per launch; 2M elems).
// hi[i] = bf16(x[i]); lo[i] = bf16(x[i] - hi[i])  => ~17-bit effective mantissa
// ---------------------------------------------------------------------------
__global__ __launch_bounds__(256) void split_bf16_kernel(
    const float* __restrict__ x, unsigned short* __restrict__ hi,
    unsigned short* __restrict__ lo, int n) {
  const int i = blockIdx.x * 256 + threadIdx.x;
  if (i < n) {
    const float f = x[i];
    const unsigned short h = f2bf_rne(f);
    hi[i] = h;
    lo[i] = f2bf_rne(f - bf2f(h));
  }
}

// ---------------------------------------------------------------------------
// Row sum-of-squares: out[row] = sum(x[row, :]^2), one 256-thread block/row
// ---------------------------------------------------------------------------
__global__ __launch_bounds__(256) void rowsumsq_kernel(const float* __restrict__ x,
                                                       float* __restrict__ out,
                                                       int d) {
  __shared__ float red[256];
  const int row = blockIdx.x;
  const float* p = x + (size_t)row * d;
  float s = 0.f;
  for (int c = threadIdx.x; c < d; c += 256) { float v = p[c]; s += v * v; }
  red[threadIdx.x] = s;
  __syncthreads();
  for (int k = 128; k > 0; k >>= 1) {
    if ((int)threadIdx.x < k) red[threadIdx.x] += red[threadIdx.x + k];
    __syncthreads();
  }
  if (threadIdx.x == 0) out[row] = red[0];
}

// ---------------------------------------------------------------------------
// Softmax over classes, one 256-thread block per sample row
// ---------------------------------------------------------------------------
__global__ __launch_bounds__(256) void softmax_kernel(const float* __restrict__ pred,
                                                      float* __restrict__ probs) {
  __shared__ float red[256];
  const int row = blockIdx.x;
  const float* p = pred + (size_t)row * CDIM;

  float mx = -3.4e38f;
  for (int c = threadIdx.x; c < CDIM; c += 256) mx = fmaxf(mx, p[c]);
  red[threadIdx.x] = mx;
  __syncthreads();
  for (int k = 128; k > 0; k >>= 1) {
    if ((int)threadIdx.x < k)
      red[threadIdx.x] = fmaxf(red[threadIdx.x], red[threadIdx.x + k]);
    __syncthreads();
  }
  mx = red[0];
  __syncthreads();

  float s = 0.f;
  for (int c = threadIdx.x; c < CDIM; c += 256) s += __expf(p[c] - mx);
  red[threadIdx.x] = s;
  __syncthreads();
  for (int k = 128; k > 0; k >>= 1) {
    if ((int)threadIdx.x < k) red[threadIdx.x] += red[threadIdx.x + k];
    __syncthreads();
  }
  const float inv = 1.0f / red[0];

  float* o = probs + (size_t)row * CDIM;
  for (int c = threadIdx.x; c < CDIM; c += 256) o[c] = __expf(p[c] - mx) * inv;
}

// ---------------------------------------------------------------------------
// GEMM + per-row min(d2) with split-bf16 WMMA, software-pipelined.
//   grid = (BDIM/128, ceil(CDIM/128)), block = 256 (8 waves)
//   wave w owns rows [bx*128 + w*16, +16) x cols [by*128, +128)
//   Pipeline: B fragments one subtile ahead (wrapping into the next K-step),
//   A raw floats one K-step ahead.
// ---------------------------------------------------------------------------
__device__ __forceinline__ void load_b(const unsigned short* __restrict__ chi,
                                       const unsigned short* __restrict__ clo,
                                       int boff, V16& bh, V16& bl) {
  const uint4* ph = (const uint4*)(chi + boff);
  const uint4* pl = (const uint4*)(clo + boff);
  bh.q[0] = ph[0]; bh.q[1] = ph[1];
  bl.q[0] = pl[0]; bl.q[1] = pl[1];
}

__device__ __forceinline__ void load_a_raw(const float* __restrict__ arow,
                                           int k0, int abase, float t[16]) {
  const float4* ap = (const float4*)(arow + k0 + 2 * abase);
  float4 x;
  x = ap[0]; t[0]=x.x;  t[1]=x.y;  t[2]=x.z;  t[3]=x.w;
  x = ap[1]; t[4]=x.x;  t[5]=x.y;  t[6]=x.z;  t[7]=x.w;
  x = ap[2]; t[8]=x.x;  t[9]=x.y;  t[10]=x.z; t[11]=x.w;   // K = base+8..11
  x = ap[3]; t[12]=x.x; t[13]=x.y; t[14]=x.z; t[15]=x.w;   // K = base+12..15
}

__global__ __launch_bounds__(256) void gemm_min_kernel(
    const float* __restrict__ features,
    const unsigned short* __restrict__ chi,   // centers bf16 hi plane [C][D]
    const unsigned short* __restrict__ clo,   // centers bf16 lo plane [C][D]
    const float* __restrict__ f2, const float* __restrict__ c2,
    unsigned int* __restrict__ min_bits) {
  const int lane  = threadIdx.x & 31;
  const int wave  = threadIdx.x >> 5;
  const int l15   = lane & 15;
  const int half  = lane >> 4;
  const int abase = half * 8;   // half-wave M offset (A K-offset is 2*abase)
  const int bbase = half * 16;  // B K-offset for this half-wave

  const int mrow0 = blockIdx.x * 128 + wave * 16;
  const int m     = mrow0 + l15;           // feature row this lane streams
  const int c0    = blockIdx.y * 128;

  const float* arow = features + (size_t)m * DDIM;

  int cidx[8], coff[8];
#pragma unroll
  for (int nt = 0; nt < 8; ++nt) {
    int c    = c0 + nt * 16 + l15;
    cidx[nt] = c;
    int ce   = (c < CDIM) ? c : (CDIM - 1);  // clamp; masked out in epilogue
    coff[nt] = ce * DDIM + bbase;            // element offset, 32B aligned
  }

  v8f acc[8];
#pragma unroll
  for (int i = 0; i < 8; ++i) acc[i] = (v8f)0.0f;

  // ---- pipeline prologue: A raw for k0=0, B fragments for (k0=0, nt=0)
  float araw[16];
  load_a_raw(arow, 0, abase, araw);
  V16 bh, bl;
  load_b(chi, clo, coff[0], bh, bl);

  for (int k0 = 0; k0 < DDIM; k0 += 32) {
    __builtin_prefetch(arow + k0 + 96, 0, 3);  // global_prefetch_b8 ahead

    // split current A chunk (overlaps with in-flight B loads / WMMAs)
    V16 ah, al;
#pragma unroll
    for (int i = 0; i < 16; ++i) {
      unsigned short h = f2bf_rne(araw[i]);
      ah.s[i] = h;
      al.s[i] = f2bf_rne(araw[i] - bf2f(h));
    }

    // issue next K-step's A loads early
    const int k0n = (k0 + 32 < DDIM) ? (k0 + 32) : 0;  // wrap (harmless reload)
    load_a_raw(arow, k0n, abase, araw);

#pragma unroll
    for (int nt = 0; nt < 8; ++nt) {
      // issue next subtile's B loads before this subtile's WMMAs
      const int ntn   = (nt + 1) & 7;
      const int knext = (nt < 7) ? k0 : k0n;
      V16 bhn, bln;
      load_b(chi, clo, coff[ntn] + knext, bhn, bln);

      acc[nt] = __builtin_amdgcn_wmma_f32_16x16x32_bf16(
          false, al.v, false, bh.v, (short)0, acc[nt], false, false);
      acc[nt] = __builtin_amdgcn_wmma_f32_16x16x32_bf16(
          false, ah.v, false, bl.v, (short)0, acc[nt], false, false);
      acc[nt] = __builtin_amdgcn_wmma_f32_16x16x32_bf16(
          false, ah.v, false, bh.v, (short)0, acc[nt], false, false);

      bh = bhn; bl = bln;
    }
  }

  // ---- Epilogue: d2 = f2 - 2*cross + c2, clamp >=0, min over this wave's
  //      128 columns, then global atomicMin (uint trick valid for d2 >= 0).
#pragma unroll
  for (int r = 0; r < 8; ++r) {
    const int   mm  = mrow0 + abase + r;  // C/D layout: lane<16 -> M=r else r+8
    const float f2v = f2[mm];
    float best = 3.0e38f;
#pragma unroll
    for (int nt = 0; nt < 8; ++nt) {
      if (cidx[nt] < CDIM) {
        float d2 = fmaf(-2.0f, acc[nt][r], f2v) + c2[cidx[nt]];
        d2 = fmaxf(d2, 0.0f);
        best = fminf(best, d2);
      }
    }
    // min across the 16 lanes that share row mm (xor masks < 16 stay in-half)
#pragma unroll
    for (int off = 1; off < 16; off <<= 1)
      best = fminf(best, __shfl_xor(best, off, 32));
    if (l15 == 0) atomicMin(min_bits + mm, __float_as_uint(best));
  }
}

// ---------------------------------------------------------------------------
// mask[b] = sqrt(min_d2[b]) / (g0 + 1.5*g1 + 1e-8) > 1
// ---------------------------------------------------------------------------
__global__ __launch_bounds__(256) void finalize_kernel(
    const unsigned int* __restrict__ min_bits,
    const float* __restrict__ gstats, float* __restrict__ mask) {
  const int i = blockIdx.x * 256 + threadIdx.x;
  if (i < BDIM) {
    const float md    = __uint_as_float(min_bits[i]);
    const float thr   = gstats[0] + 1.5f * gstats[1] + 1e-8f;
    const float score = sqrtf(md) / thr;
    mask[i] = (score > 1.0f) ? 1.0f : 0.0f;
  }
}

// ---------------------------------------------------------------------------
// launch
// ---------------------------------------------------------------------------
extern "C" void kernel_launch(void* const* d_in, const int* in_sizes, int n_in,
                              void* d_out, int out_size, void* d_ws, size_t ws_size,
                              hipStream_t stream) {
  (void)in_sizes; (void)n_in; (void)out_size; (void)ws_size;

  const float* features    = (const float*)d_in[0];  // [B, D] f32
  const float* predictions = (const float*)d_in[1];  // [B, C] f32
  const float* centers     = (const float*)d_in[2];  // [C, D] f32
  const float* gstats      = (const float*)d_in[3];  // [2]    f32
  // d_in[4] known_labels (int64), d_in[5] num_known_classes: unused

  float* mask_out  = (float*)d_out;               // [B] (bool as 0/1 float)
  float* probs_out = (float*)d_out + BDIM;        // [B, C] f32

  // Workspace layout (~8.4 MB total)
  char* ws = (char*)d_ws;
  unsigned int*   min_bits = (unsigned int*)ws;                    // 64 KB
  float*          c2       = (float*)(ws + 65536);                 // 4 KB slot
  float*          f2       = (float*)(ws + 69632);                 // 64 KB
  unsigned short* chi      = (unsigned short*)(ws + 135168);       // 4.0 MB
  unsigned short* clo      = (unsigned short*)(ws + 135168 + (size_t)CDIM * DDIM * 2);

  // 0x7F7F7F7F ~= 3.39e38 acts as +inf for squared distances
  hipMemsetAsync(min_bits, 0x7F, (size_t)BDIM * sizeof(unsigned int), stream);

  const int ncd = CDIM * DDIM;
  split_bf16_kernel<<<(ncd + 255) / 256, 256, 0, stream>>>(centers, chi, clo, ncd);
  rowsumsq_kernel<<<CDIM, 256, 0, stream>>>(centers, c2, DDIM);
  rowsumsq_kernel<<<BDIM, 256, 0, stream>>>(features, f2, DDIM);
  softmax_kernel<<<BDIM, 256, 0, stream>>>(predictions, probs_out);

  dim3 grid(BDIM / 128, (CDIM + 127) / 128);  // (128, 8)
  gemm_min_kernel<<<grid, 256, 0, stream>>>(features, chi, clo, f2, c2, min_bits);

  finalize_kernel<<<(BDIM + 255) / 256, 256, 0, stream>>>(min_bits, gstats, mask_out);
}
